// Pattention_65524021067985
// MI455X (gfx1250) — matrix operand order repaired and checked by reading, not compiled, not run on hardware
//
#include <hip/hip_runtime.h>
#include <hip/hip_bf16.h>
#include <stdint.h>

// ---------------------------------------------------------------------------
// Problem constants (from the reference):
//   B=4, L=2048  -> M = 8192 rows
//   D1 = 1024 (query/key depth), N = 4096 (param tokens), D2 = 1024 (out depth)
// ---------------------------------------------------------------------------
#define M_TOTAL 8192
#define N_TOTAL 4096
#define D1_DIM  1024
#define D2_DIM  1024

#define TILE_K  32
#define PITCH   36   // LDS row pitch in bf16 elements (72B; 18-bank stride, conflict-free)

typedef __attribute__((ext_vector_type(16))) __bf16 v16bf;
typedef __attribute__((ext_vector_type(8)))  float  v8f;

union Frag16 { uint32_t u[8]; v16bf v; };

// Single-instruction packed fp32 -> 2x bf16 (RNE) : v_cvt_pk_bf16_f32
static __device__ __forceinline__ uint32_t pack2bf(float a, float b) {
  uint32_t r;
  asm("v_cvt_pk_bf16_f32 %0, %1, %2" : "=v"(r) : "v"(a), "v"(b));
  return r;
}
static __device__ __forceinline__ uint16_t f2bf(float f) {
  return (uint16_t)pack2bf(f, 0.0f);
}
static __device__ __forceinline__ float bf2f_lo(uint32_t u) {
  union { uint32_t u; float f; } x; x.u = (u << 16); return x.f;
}
static __device__ __forceinline__ float bf2f_hi(uint32_t u) {
  union { uint32_t u; float f; } x; x.u = (u & 0xFFFF0000u); return x.f;
}
static __device__ __forceinline__ float gelu_exact(float x) {
  return 0.5f * x * (1.0f + erff(x * 0.70710678118654752440f));
}

// ---------------------------------------------------------------------------
// Kernel 1: scores[m][n] = sum_d X[m][d] * K[n][d]   (bf16 out, fp32 accum)
// Block: 128(M) x 128(N), 256 threads = 8 waves in a 2(M) x 4(N) grid.
// Each wave owns 4 M-tiles x 2 N-tiles of 16x16 -> 8 accumulators.
// ---------------------------------------------------------------------------
__global__ __launch_bounds__(256) void gemm1_scores(
    const float* __restrict__ X,     // [8192][1024]
    const float* __restrict__ Kp,    // [4096][1024]
    uint16_t* __restrict__ S)        // [8192][4096] bf16
{
  __shared__ uint16_t ldsA[128 * PITCH];
  __shared__ uint16_t ldsB[128 * PITCH];

  const int t    = threadIdx.x;
  const int lane = t & 31;
  const int wave = t >> 5;
  const int wm   = wave >> 2;          // 0..1
  const int wn   = wave & 3;           // 0..3
  const int hi   = (lane >> 4) & 1;
  const int l16  = lane & 15;

  const int mBlock = blockIdx.y * 128;
  const int nBlock = blockIdx.x * 128;

  // staging assignment: thread -> (row, 16-element half-row)
  const int sRow = t >> 1;             // 0..127
  const int sCol = (t & 1) * 16;       // 0 or 16

  // per-lane K element offsets inside a 32-wide LDS row (A and B operands)
  int kaOff[8], kbOff[8];
#pragma unroll
  for (int v = 0; v < 8; ++v) {
    kaOff[v] = ((v < 4) ? 0 : 16) + hi * 8 + 2 * (v & 3);  // A: 16x32 bf16 layout
    kbOff[v] = ((v < 4) ? 0 : 16) + hi * 8 + 2 * (v & 3);  // B: 32x16 bf16 layout
  }

  v8f acc[4][2];
#pragma unroll
  for (int mi = 0; mi < 4; ++mi)
#pragma unroll
    for (int ni = 0; ni < 2; ++ni)
      acc[mi][ni] = (v8f){0.f, 0.f, 0.f, 0.f, 0.f, 0.f, 0.f, 0.f};

  for (int k0 = 0; k0 < D1_DIM; k0 += TILE_K) {
    // ---- stage A tile: X[mBlock+r][k0 .. k0+31] -> ldsA[r][k] (fp32 -> bf16)
    {
      const float4* src =
          reinterpret_cast<const float4*>(X + (size_t)(mBlock + sRow) * D1_DIM + k0 + sCol);
      uint32_t* dst = reinterpret_cast<uint32_t*>(ldsA + sRow * PITCH + sCol);
#pragma unroll
      for (int i = 0; i < 4; ++i) {
        float4 v = src[i];
        dst[i * 2 + 0] = pack2bf(v.x, v.y);
        dst[i * 2 + 1] = pack2bf(v.z, v.w);
      }
    }
    // ---- stage B tile: Kp[nBlock+r][k0 .. k0+31] -> ldsB[r][k]
    {
      const float4* src =
          reinterpret_cast<const float4*>(Kp + (size_t)(nBlock + sRow) * D1_DIM + k0 + sCol);
      uint32_t* dst = reinterpret_cast<uint32_t*>(ldsB + sRow * PITCH + sCol);
#pragma unroll
      for (int i = 0; i < 4; ++i) {
        float4 v = src[i];
        dst[i * 2 + 0] = pack2bf(v.x, v.y);
        dst[i * 2 + 1] = pack2bf(v.z, v.w);
      }
    }
    if (k0 + TILE_K < D1_DIM) {   // L2 prefetch of next tiles (global_prefetch_b8)
      __builtin_prefetch(X  + (size_t)(mBlock + sRow) * D1_DIM + k0 + TILE_K + sCol, 0, 1);
      __builtin_prefetch(Kp + (size_t)(nBlock + sRow) * D1_DIM + k0 + TILE_K + sCol, 0, 1);
    }
    __syncthreads();

    // ---- build fragments from LDS
    Frag16 af[4], bfr[2];
#pragma unroll
    for (int mi = 0; mi < 4; ++mi) {
      const uint16_t* ap = ldsA + (wm * 64 + mi * 16 + l16) * PITCH;
#pragma unroll
      for (int v = 0; v < 8; ++v)
        af[mi].u[v] = *reinterpret_cast<const uint32_t*>(ap + kaOff[v]);
    }
#pragma unroll
    for (int ni = 0; ni < 2; ++ni) {
      const uint16_t* bp = ldsB + (wn * 32 + ni * 16 + l16) * PITCH;
#pragma unroll
      for (int v = 0; v < 8; ++v)
        bfr[ni].u[v] = *reinterpret_cast<const uint32_t*>(bp + kbOff[v]);
    }

    // ---- 8 WMMAs
#pragma unroll
    for (int mi = 0; mi < 4; ++mi)
#pragma unroll
      for (int ni = 0; ni < 2; ++ni)
        acc[mi][ni] = __builtin_amdgcn_wmma_f32_16x16x32_bf16(
            false, af[mi].v, false, bfr[ni].v, (short)0, acc[mi][ni], false, false);

    __syncthreads();
  }

  // ---- epilogue: C/D layout -> scores (bf16)
#pragma unroll
  for (int mi = 0; mi < 4; ++mi)
#pragma unroll
    for (int ni = 0; ni < 2; ++ni) {
      const int n = nBlock + wn * 32 + ni * 16 + l16;
#pragma unroll
      for (int v = 0; v < 8; ++v) {
        const int m = mBlock + wm * 64 + mi * 16 + hi * 8 + v;
        S[(size_t)m * N_TOTAL + n] = f2bf(acc[mi][ni][v]);
      }
    }
}

// ---------------------------------------------------------------------------
// Kernel 2: per-row L2 norm over N=4096, rescale by sqrt(N)=64, exact GELU.
// One 256-thread block per row; 16 bf16 elements per thread; in-place rewrite.
// ---------------------------------------------------------------------------
__global__ __launch_bounds__(256) void rownorm_gelu(uint16_t* __restrict__ S)
{
  const int t = threadIdx.x;
  uint16_t* rp = S + (size_t)blockIdx.x * N_TOTAL;

  const uint4* src = reinterpret_cast<const uint4*>(rp) + (size_t)t * 2;
  uint4 u0 = src[0];
  uint4 u1 = src[1];

  float vals[16];
  {
    uint32_t w[8] = {u0.x, u0.y, u0.z, u0.w, u1.x, u1.y, u1.z, u1.w};
#pragma unroll
    for (int i = 0; i < 8; ++i) {
      vals[2 * i + 0] = bf2f_lo(w[i]);
      vals[2 * i + 1] = bf2f_hi(w[i]);
    }
  }

  float ss = 0.f;
#pragma unroll
  for (int i = 0; i < 16; ++i) ss += vals[i] * vals[i];

  __shared__ float red[256];
  red[t] = ss;
  __syncthreads();
#pragma unroll
  for (int s = 128; s > 0; s >>= 1) {
    if (t < s) red[t] += red[t + s];
    __syncthreads();
  }
  const float scale = 64.0f / sqrtf(red[0]);   // sqrt(4096) / l2

  uint32_t out[8];
#pragma unroll
  for (int i = 0; i < 8; ++i) {
    const float g0 = gelu_exact(vals[2 * i + 0] * scale);
    const float g1 = gelu_exact(vals[2 * i + 1] * scale);
    out[i] = pack2bf(g0, g1);
  }
  uint4* dst = reinterpret_cast<uint4*>(rp) + (size_t)t * 2;
  dst[0] = (uint4){out[0], out[1], out[2], out[3]};
  dst[1] = (uint4){out[4], out[5], out[6], out[7]};
}

// ---------------------------------------------------------------------------
// Kernel 3: out[m][n] = sum_k G[m][k] * V[k][n]   (fp32 out)
// G is the gelu'd score matrix (bf16, [8192][4096]); V is fp32 [4096][1024].
// A tile is raw bf16 -> moved with GLOBAL_LOAD_ASYNC_TO_LDS_B128 (ASYNCcnt),
// no VGPR round-trip. V is converted+transposed through VGPRs in parallel.
// ---------------------------------------------------------------------------
__global__ __launch_bounds__(256) void gemm2_out(
    const uint16_t* __restrict__ G,  // [8192][4096] bf16
    const float* __restrict__ Vp,    // [4096][1024]
    float* __restrict__ O)           // [8192][1024]
{
  __shared__ uint16_t ldsA[128 * PITCH];
  __shared__ uint16_t ldsB[128 * PITCH];

  const int t    = threadIdx.x;
  const int lane = t & 31;
  const int wave = t >> 5;
  const int wm   = wave >> 2;
  const int wn   = wave & 3;
  const int hi   = (lane >> 4) & 1;
  const int l16  = lane & 15;

  const int mBlock = blockIdx.y * 128;
  const int nBlock = blockIdx.x * 128;

  const int sRow = t >> 1;
  const int sCol = (t & 1) * 16;
  const int bk   = t >> 3;            // 0..31  (V source row within K-tile)
  const int bn0  = (t & 7) * 16;      // 0..112 (V source column start)

  // per-lane LDS byte address for the async A copy (addrspace(3) low 32 bits)
  const uint32_t ldsAdst = (uint32_t)(uintptr_t)(ldsA + sRow * PITCH + sCol);

  int kaOff[8], kbOff[8];
#pragma unroll
  for (int v = 0; v < 8; ++v) {
    kaOff[v] = ((v < 4) ? 0 : 16) + hi * 8 + 2 * (v & 3);
    kbOff[v] = ((v < 4) ? 0 : 16) + hi * 8 + 2 * (v & 3);
  }

  v8f acc[4][2];
#pragma unroll
  for (int mi = 0; mi < 4; ++mi)
#pragma unroll
    for (int ni = 0; ni < 2; ++ni)
      acc[mi][ni] = (v8f){0.f, 0.f, 0.f, 0.f, 0.f, 0.f, 0.f, 0.f};

  for (int k0 = 0; k0 < N_TOTAL; k0 += TILE_K) {
    // ---- stage A tile: G[mBlock+r][k0 .. k0+31] (bf16) -> ldsA[r][k]
    //      async copy, 2 x 16B per thread, tracked by ASYNCcnt
    {
      const uint64_t gaddr =
          (uint64_t)(uintptr_t)(G + (size_t)(mBlock + sRow) * N_TOTAL + k0 + sCol);
      asm volatile(
          "global_load_async_to_lds_b128 %0, %1, off\n\t"
          "global_load_async_to_lds_b128 %0, %1, off offset:16"
          :
          : "v"(ldsAdst), "v"(gaddr)
          : "memory");
    }
    // ---- stage B tile with transpose: Vp[k0+bk][nBlock + bn0 .. +15] -> ldsB[n][k]
    {
      const float4* src =
          reinterpret_cast<const float4*>(Vp + (size_t)(k0 + bk) * D2_DIM + nBlock + bn0);
#pragma unroll
      for (int i = 0; i < 4; ++i) {
        float4 v = src[i];
        ldsB[(bn0 + i * 4 + 0) * PITCH + bk] = f2bf(v.x);
        ldsB[(bn0 + i * 4 + 1) * PITCH + bk] = f2bf(v.y);
        ldsB[(bn0 + i * 4 + 2) * PITCH + bk] = f2bf(v.z);
        ldsB[(bn0 + i * 4 + 3) * PITCH + bk] = f2bf(v.w);
      }
    }
    if (k0 + TILE_K < N_TOTAL)
      __builtin_prefetch(Vp + (size_t)(k0 + TILE_K + bk) * D2_DIM + nBlock + bn0, 0, 1);

    asm volatile("s_wait_asynccnt 0x0" ::: "memory");  // A tile landed in LDS
    __syncthreads();

    Frag16 af[4], bfr[2];
#pragma unroll
    for (int mi = 0; mi < 4; ++mi) {
      const uint16_t* ap = ldsA + (wm * 64 + mi * 16 + l16) * PITCH;
#pragma unroll
      for (int v = 0; v < 8; ++v)
        af[mi].u[v] = *reinterpret_cast<const uint32_t*>(ap + kaOff[v]);
    }
#pragma unroll
    for (int ni = 0; ni < 2; ++ni) {
      const uint16_t* bp = ldsB + (wn * 32 + ni * 16 + l16) * PITCH;
#pragma unroll
      for (int v = 0; v < 8; ++v)
        bfr[ni].u[v] = *reinterpret_cast<const uint32_t*>(bp + kbOff[v]);
    }

#pragma unroll
    for (int mi = 0; mi < 4; ++mi)
#pragma unroll
      for (int ni = 0; ni < 2; ++ni)
        acc[mi][ni] = __builtin_amdgcn_wmma_f32_16x16x32_bf16(
            false, af[mi].v, false, bfr[ni].v, (short)0, acc[mi][ni], false, false);

    __syncthreads();
  }

  // ---- epilogue: fp32 output
#pragma unroll
  for (int mi = 0; mi < 4; ++mi)
#pragma unroll
    for (int ni = 0; ni < 2; ++ni) {
      const int n = nBlock + wn * 32 + ni * 16 + l16;
#pragma unroll
      for (int v = 0; v < 8; ++v) {
        const int m = mBlock + wm * 64 + mi * 16 + hi * 8 + v;
        O[(size_t)m * D2_DIM + n] = acc[mi][ni][v];
      }
    }
}

// ---------------------------------------------------------------------------
// Launch: inputs = { inputs fp32 [4,2048,1024], key fp32 [4096,1024],
//                    value fp32 [4096,1024] }, out = fp32 [4,2048,1024].
// Workspace: 8192*4096 bf16 scores = 64 MB in d_ws.
// ---------------------------------------------------------------------------
extern "C" void kernel_launch(void* const* d_in, const int* in_sizes, int n_in,
                              void* d_out, int out_size, void* d_ws, size_t ws_size,
                              hipStream_t stream) {
  (void)in_sizes; (void)n_in; (void)out_size; (void)ws_size;
  const float* X  = (const float*)d_in[0];
  const float* Kp = (const float*)d_in[1];
  const float* Vp = (const float*)d_in[2];
  float*    O = (float*)d_out;
  uint16_t* S = (uint16_t*)d_ws;   // 8192*4096*2 bytes

  gemm1_scores<<<dim3(N_TOTAL / 128, M_TOTAL / 128), 256, 0, stream>>>(X, Kp, S);
  rownorm_gelu<<<dim3(M_TOTAL), 256, 0, stream>>>(S);
  gemm2_out<<<dim3(D2_DIM / 128, M_TOTAL / 128), 256, 0, stream>>>(S, Vp, O);
}
